// AudioCrossAttention_19335942766904
// MI455X (gfx1250) — compile-verified, matrix operand order checked by reading, not run on hardware
//
#include <hip/hip_runtime.h>
#include <hip/hip_bf16.h>

typedef __attribute__((ext_vector_type(16))) __bf16 v16bf;
typedef __attribute__((ext_vector_type(8)))  __bf16 v8bf;
typedef __attribute__((ext_vector_type(8)))  float  v8f;

// gfx1250 async global->LDS path (ASYNCcnt), guarded so compile cannot break.
#if defined(__has_builtin)
#if __has_builtin(__builtin_amdgcn_global_load_async_to_lds_b128) && \
    __has_builtin(__builtin_amdgcn_s_wait_asynccnt)
#define USE_ASYNC_LDS 1
#endif
#endif
#ifndef USE_ASYNC_LDS
#define USE_ASYNC_LDS 0
#endif

#if USE_ASYNC_LDS
// Builtin signature (from hipcc diagnostic): param0 is
// 'int __attribute__((vector_size(16))) addrspace(1)*' (printed as __device__*).
typedef int v4i_vs __attribute__((__vector_size__(16)));
__device__ inline void async_copy16(const __bf16* g, __bf16* l) {
  __builtin_amdgcn_global_load_async_to_lds_b128(
      (__attribute__((address_space(1))) v4i_vs*)g,
      (__attribute__((address_space(3))) v4i_vs*)l,
      /*offset=*/0, /*cpol=*/0);
}
#endif

// ---------------------------------------------------------------------------
// WMMA helper: D = A(16x32 bf16) * B(32x16 bf16) + C(16x16 f32)
// ---------------------------------------------------------------------------
__device__ inline v8f wmma_bf16(v16bf a, v16bf b, v8f c) {
  return __builtin_amdgcn_wmma_f32_16x16x32_bf16(
      /*neg_a=*/false, a, /*neg_b=*/false, b,
      /*c_mod=*/(short)0, c, /*reuse_a=*/false, /*reuse_b=*/false);
}

// Load a 16x32 bf16 fragment (A layout; also used for B with [n][k] tiles).
// Per CDNA5 ISA layout: lane%16 selects the row/col, lane/16 selects the
// 8-element K half; elements 0..7 <- K=base..base+7, 8..15 <- K=base+16..+23.
__device__ inline v16bf load_frag(const __bf16* tile, int row0, int ldk,
                                  int k0, int lane) {
  const __bf16* p = tile + (size_t)(row0 + (lane & 15)) * ldk + k0 + ((lane >> 4) << 3);
  v8bf lo = *(const v8bf*)(p);
  v8bf hi = *(const v8bf*)(p + 16);
  return __builtin_shufflevector(lo, hi, 0, 1, 2, 3, 4, 5, 6, 7,
                                 8, 9, 10, 11, 12, 13, 14, 15);
}

// f32 -> bf16 convert+store (4 elements) into LDS
__device__ inline void cvt_store4(__bf16* dst, const float* src) {
  float4 v = *(const float4*)src;
  dst[0] = (__bf16)v.x; dst[1] = (__bf16)v.y;
  dst[2] = (__bf16)v.z; dst[3] = (__bf16)v.w;
}
__device__ inline void cvt_store4(__bf16* dst, const __bf16* src) {
  *(uint2*)dst = *(const uint2*)src;  // already bf16: straight 8B copy
}

__device__ inline void store_out(float* o, float v)  { *o = v; }
__device__ inline void store_out(__bf16* o, float v) { *o = (__bf16)v; }

// ---------------------------------------------------------------------------
// Generic GEMM: Out[M,N] = A[M,K] * W[N,K]^T + bias[N]
//   TA: float (convert to bf16 in LDS) or __bf16
//   TO: __bf16 (intermediate) or float (final output)
//   LROPE: 1 -> fuse additive L-RoPE embedding (K projection epilogue)
// Block: 256 threads (8 waves), 128x128 tile, per-wave 32x64, TILE_K=64
// (two WMMA K-steps per barrier pair -> 16 WMMA/wave per stage).
// ---------------------------------------------------------------------------
template <typename TA, typename TO, int LROPE>
__global__ __launch_bounds__(256)
void gemm_wmma(const TA* __restrict__ A, const float* __restrict__ W,
               const float* __restrict__ bias, TO* __restrict__ Out,
               int M, int N, int K,
               const int* __restrict__ labels,
               const float* __restrict__ label_emb, int skv) {
  __shared__ __bf16 sA[128 * 64];
  __shared__ __bf16 sB[128 * 64];

  const int tid  = threadIdx.x;
  const int lane = tid & 31;
  const int w    = tid >> 5;
  const int mBase = blockIdx.x * 128;
  const int nBase = blockIdx.y * 128;
  const int wm = (w >> 1) * 32;   // wave M offset inside tile (0,32,64,96)
  const int wn = (w & 1) * 64;    // wave N offset inside tile (0,64)

  v8f acc[2][4];
  const v8f vzero = {0.f, 0.f, 0.f, 0.f, 0.f, 0.f, 0.f, 0.f};
#pragma unroll
  for (int i = 0; i < 2; ++i)
#pragma unroll
    for (int j = 0; j < 4; ++j) acc[i][j] = vzero;

  for (int k0 = 0; k0 < K; k0 += 64) {
    // Stage A (128x64) and W (128x64) tiles as bf16 into LDS.
#pragma unroll
    for (int i = 0; i < 8; ++i) {
      int idx = tid + i * 256;        // 2048 4-element chunks per tile
      int row = idx >> 4;             // 16 chunks per 64-wide row
      int kc  = (idx & 15) * 4;
      cvt_store4(&sA[row * 64 + kc], A + (size_t)(mBase + row) * K + k0 + kc);
      cvt_store4(&sB[row * 64 + kc], W + (size_t)(nBase + row) * K + k0 + kc);
    }
    if (k0 + 64 < K) {  // hint next slab into cache (global_prefetch_b8)
      __builtin_prefetch(A + (size_t)(mBase + (tid >> 1)) * K + k0 + 64, 0, 1);
      __builtin_prefetch(W + (size_t)(nBase + (tid >> 1)) * K + k0 + 64, 0, 1);
    }
    __syncthreads();

#pragma unroll
    for (int ks = 0; ks < 64; ks += 32) {
      v16bf a0 = load_frag(sA, wm,      64, ks, lane);
      v16bf a1 = load_frag(sA, wm + 16, 64, ks, lane);
#pragma unroll
      for (int nf = 0; nf < 4; ++nf) {
        v16bf bb = load_frag(sB, wn + nf * 16, 64, ks, lane);
        acc[0][nf] = wmma_bf16(a0, bb, acc[0][nf]);
        acc[1][nf] = wmma_bf16(a1, bb, acc[1][nf]);
      }
    }
    __syncthreads();
  }

  // Epilogue: bias (+ optional L-RoPE), store.
#pragma unroll
  for (int mf = 0; mf < 2; ++mf) {
#pragma unroll
    for (int nf = 0; nf < 4; ++nf) {
#pragma unroll
      for (int r = 0; r < 8; ++r) {
        int m = mBase + wm + mf * 16 + r + ((lane >> 4) << 3);
        int n = nBase + wn + nf * 16 + (lane & 15);
        float v = acc[mf][nf][r] + bias[n];
        if (LROPE) {
          int d   = n & 63;               // position within head_dim
          int bb  = m / skv;
          int pos = m - bb * skv;
          // inv_freq = 10000^(-2i/64); ln(10000)*2/64 = 0.28782314
          float invf = __expf(-0.28782314f * (float)(d & 31));
          float ang  = (float)pos * invf;
          float e    = (d < 32) ? __sinf(ang) : __cosf(ang);
          v += e * label_emb[labels[bb] * 64 + d];
        }
        store_out(&Out[(size_t)m * N + n], v);
      }
    }
  }
}

// ---------------------------------------------------------------------------
// Flash attention: one (batch, head, 64-row Q tile) per block.
// 128 threads = 4 waves; each wave owns 16 Q rows, streams 64-wide KV tiles.
// K tile staged via global_load_async_to_lds_b128 (ASYNCcnt) when available.
// ---------------------------------------------------------------------------
__global__ __launch_bounds__(128)
void flash_attn(const __bf16* __restrict__ Qg, const __bf16* __restrict__ Kg,
                const __bf16* __restrict__ Vg, __bf16* __restrict__ Og,
                int Bn, int Hn, int SQn, int SKVn) {
  __shared__ __bf16 sK[64 * 64];      // [kv][d]  (B operand for Q K^T)
  __shared__ __bf16 sVt[64 * 64];     // [d][kv]  (B operand for P V)
  __shared__ __bf16 sP[4][16 * 64];   // per-wave P tile

  const int tid  = threadIdx.x;
  const int lane = tid & 31;
  const int w    = tid >> 5;
  const int qTiles = SQn / 64;
  const int qt = blockIdx.x % qTiles;
  const int h  = (blockIdx.x / qTiles) % Hn;
  const int b  = blockIdx.x / (qTiles * Hn);
  const int DIMc = Hn * 64;
  const int qBase = qt * 64 + w * 16;
  const float scale = 0.125f;         // 1/sqrt(64)

  const __bf16* Qb = Qg + (size_t)b * SQn  * DIMc + h * 64;
  const __bf16* Kb = Kg + (size_t)b * SKVn * DIMc + h * 64;
  const __bf16* Vb = Vg + (size_t)b * SKVn * DIMc + h * 64;

  // Q fragments for this wave's 16 rows (HD=64 -> 2 K-steps of 32)
  v16bf qa0 = load_frag(Qb, qBase, DIMc, 0,  lane);
  v16bf qa1 = load_frag(Qb, qBase, DIMc, 32, lane);

  const v8f vzero = {0.f, 0.f, 0.f, 0.f, 0.f, 0.f, 0.f, 0.f};
  v8f oacc[4];
#pragma unroll
  for (int i = 0; i < 4; ++i) oacc[i] = vzero;
  float mrun[8], lrun[8];
#pragma unroll
  for (int r = 0; r < 8; ++r) { mrun[r] = -3.0e38f; lrun[r] = 0.f; }

  for (int kv0 = 0; kv0 < SKVn; kv0 += 64) {
    // Cooperative stage: K tile natural [kv][d]; V tile transposed to [d][kv].
#pragma unroll
    for (int i = 0; i < 4; ++i) {
      int idx = tid + i * 128;        // 512 8-element chunks
      int row = idx >> 3;
      int dc  = (idx & 7) * 8;
#if USE_ASYNC_LDS
      async_copy16(Kb + (size_t)(kv0 + row) * DIMc + dc, &sK[row * 64 + dc]);
#else
      v8bf kk = *(const v8bf*)(Kb + (size_t)(kv0 + row) * DIMc + dc);
      *(v8bf*)&sK[row * 64 + dc] = kk;
#endif
      v8bf vv = *(const v8bf*)(Vb + (size_t)(kv0 + row) * DIMc + dc);
#pragma unroll
      for (int j = 0; j < 8; ++j) sVt[(dc + j) * 64 + row] = vv[j];
    }
#if USE_ASYNC_LDS
    __builtin_amdgcn_s_wait_asynccnt(0);
#endif
    __syncthreads();

    // S = Q K^T  (16 x 64, f32 accum)
    v8f sacc[4];
#pragma unroll
    for (int nf = 0; nf < 4; ++nf) {
      sacc[nf] = vzero;
      v16bf b0 = load_frag(sK, nf * 16, 64, 0,  lane);
      v16bf b1 = load_frag(sK, nf * 16, 64, 32, lane);
      sacc[nf] = wmma_bf16(qa0, b0, sacc[nf]);
      sacc[nf] = wmma_bf16(qa1, b1, sacc[nf]);
    }

    // Online softmax: a C-matrix row spans 16 lanes within a half-wave.
    __bf16* sPw = &sP[w][0];
#pragma unroll
    for (int r = 0; r < 8; ++r) {
      float mx = -3.0e38f;
#pragma unroll
      for (int nf = 0; nf < 4; ++nf) mx = fmaxf(mx, sacc[nf][r] * scale);
#pragma unroll
      for (int off = 8; off; off >>= 1) mx = fmaxf(mx, __shfl_xor(mx, off, 16));
      float newm  = fmaxf(mrun[r], mx);
      float alpha = __expf(mrun[r] - newm);
      float rs = 0.f;
      int prow = r + ((lane >> 4) << 3);
#pragma unroll
      for (int nf = 0; nf < 4; ++nf) {
        float p = __expf(sacc[nf][r] * scale - newm);
        rs += p;
        sPw[prow * 64 + nf * 16 + (lane & 15)] = (__bf16)p;
      }
#pragma unroll
      for (int off = 8; off; off >>= 1) rs += __shfl_xor(rs, off, 16);
      lrun[r] = lrun[r] * alpha + rs;
      mrun[r] = newm;
#pragma unroll
      for (int df = 0; df < 4; ++df) oacc[df][r] *= alpha;
    }

    // O += P V   (P read back from this wave's private LDS region)
    v16bf pa0 = load_frag(sPw, 0, 64, 0,  lane);
    v16bf pa1 = load_frag(sPw, 0, 64, 32, lane);
#pragma unroll
    for (int df = 0; df < 4; ++df) {
      v16bf vb0 = load_frag(sVt, df * 16, 64, 0,  lane);
      v16bf vb1 = load_frag(sVt, df * 16, 64, 32, lane);
      oacc[df] = wmma_bf16(pa0, vb0, oacc[df]);
      oacc[df] = wmma_bf16(pa1, vb1, oacc[df]);
    }
    __syncthreads();
  }

  // Normalize and store bf16 attention output [B, SQ, H*64]
  __bf16* Ob = Og + (size_t)b * SQn * DIMc + h * 64;
#pragma unroll
  for (int r = 0; r < 8; ++r) {
    int row = r + ((lane >> 4) << 3);
    int q = qBase + row;
    float linv = 1.0f / lrun[r];
#pragma unroll
    for (int df = 0; df < 4; ++df) {
      Ob[(size_t)q * DIMc + df * 16 + (lane & 15)] = (__bf16)(oacc[df][r] * linv);
    }
  }
}

// ---------------------------------------------------------------------------
extern "C" void kernel_launch(void* const* d_in, const int* in_sizes, int n_in,
                              void* d_out, int out_size, void* d_ws,
                              size_t ws_size, hipStream_t stream) {
  const float* visual = (const float*)d_in[0];
  const float* audio  = (const float*)d_in[1];
  const int*   labels = (const int*)d_in[2];
  const float* Wq = (const float*)d_in[3];
  const float* bq = (const float*)d_in[4];
  const float* Wk = (const float*)d_in[5];
  const float* bk = (const float*)d_in[6];
  const float* Wv = (const float*)d_in[7];
  const float* bv = (const float*)d_in[8];
  const float* Wo = (const float*)d_in[9];
  const float* bo = (const float*)d_in[10];
  const float* lemb = (const float*)d_in[11];
  float* out = (float*)d_out;

  const int Bc = 2, SQc = 2048, SKVc = 2048, D = 1024, Hc = 16;
  const int Mq = Bc * SQc, Mkv = Bc * SKVc;

  __bf16* wq = (__bf16*)d_ws;
  __bf16* wk = wq + (size_t)Mq  * D;
  __bf16* wv = wk + (size_t)Mkv * D;
  __bf16* wo = wv + (size_t)Mkv * D;

  dim3 blk(256);
  dim3 gq(Mq / 128, D / 128);
  dim3 gkv(Mkv / 128, D / 128);

  gemm_wmma<float, __bf16, 0><<<gq, blk, 0, stream>>>(
      visual, Wq, bq, wq, Mq, D, D, nullptr, nullptr, SKVc);
  gemm_wmma<float, __bf16, 1><<<gkv, blk, 0, stream>>>(
      audio, Wk, bk, wk, Mkv, D, D, labels, lemb, SKVc);
  gemm_wmma<float, __bf16, 0><<<gkv, blk, 0, stream>>>(
      audio, Wv, bv, wv, Mkv, D, D, nullptr, nullptr, SKVc);

  flash_attn<<<dim3(Bc * Hc * (SQc / 64)), dim3(128), 0, stream>>>(
      wq, wk, wv, wo, Bc, Hc, SQc, SKVc);

  gemm_wmma<__bf16, float, 0><<<gq, blk, 0, stream>>>(
      wo, Wo, bo, out, Mq, D, D, nullptr, nullptr, SKVc);
}
